// HierarchicalReasoningModel_ACTV1_Inner_30176440221703
// MI455X (gfx1250) — compile-verified, matrix operand order checked by reading, not run on hardware
//
#include <hip/hip_runtime.h>
#include <hip/hip_bf16.h>

#define Hdim    256
#define NEXP    160
#define TOPK    6
#define NDEV    8
#define EPD_    20
#define MAXDEV_ 3
#define INTER_  768
#define TTOT    512
#define TILE_M  32         // tokens per block tile (2 WMMA M-fragments)
#define LDX     264        // 256 + pad (row stride of s_x, elements)
#define LDA     776        // 768 + pad (row stride of s_act, elements)

typedef __attribute__((ext_vector_type(16))) __bf16 v16bf;
typedef __attribute__((ext_vector_type(8)))  __bf16 v8bf;
typedef __attribute__((ext_vector_type(4)))  __bf16 v4bf;
typedef __attribute__((ext_vector_type(8)))  float  v8f;

__device__ __forceinline__ v8f wmma_bf16(v16bf a, v16bf b, v8f c) {
  // D = A(16x32 bf16) * B(32x16 bf16) + C(16x16 f32)
  return __builtin_amdgcn_wmma_f32_16x16x32_bf16(false, a, false, b, (short)0, c,
                                                 false, false);
}

// A fragment: 16(M) x 32(K) bf16 from row-major LDS.
// Per-lane data is two contiguous 16B runs -> two ds_load_b128.
__device__ __forceinline__ v16bf frag_A(const __bf16* base, int ld, int lane) {
  const __bf16* p = base + (lane & 15) * ld + ((lane >> 4) << 3);
  v8bf lo = *(const v8bf*)p;
  v8bf hi = *(const v8bf*)(p + 16);
  return __builtin_shufflevector(lo, hi, 0, 1, 2, 3, 4, 5, 6, 7,
                                 8, 9, 10, 11, 12, 13, 14, 15);
}

// B fragment: 32(K) x 16(N) f32 weight tile loaded DIRECTLY from global and
// converted to bf16 in registers. For each element slot j the 32 lanes read
// two contiguous 64B rows -> coalesced. No LDS round-trip for weights.
__device__ __forceinline__ v16bf load_B_global(const float* __restrict__ src,
                                               int ldsrc, int lane) {
  const int n  = lane & 15;
  const int kb = (lane >> 4) << 3;
  const float* p = src + n;
  v16bf b;
#pragma unroll
  for (int j = 0; j < 8; ++j) b[j] = (__bf16)p[(kb + j) * ldsrc];
#pragma unroll
  for (int j = 0; j < 8; ++j) b[8 + j] = (__bf16)p[(16 + kb + j) * ldsrc];
  return b;
}

// One 32-token tile through a SwiGLU expert. Block = 128 threads = 4 waves.
// Every B fragment (weight tile) is reused by two M-fragments, halving the
// dominant HBM weight traffic vs a 16-token tile.
__device__ void expert_tile(const float* __restrict__ x,
                            const float* __restrict__ Wgu_e,   // [256 x 1536]
                            const float* __restrict__ Wd_e,    // [768 x 256]
                            float* __restrict__ out,
                            const int* s_tok, const float* s_w,
                            __bf16* s_x, __bf16* s_act,
                            int tid, bool use_atomic) {
  const int lane = tid & 31;
  const int wave = tid >> 5;

  __syncthreads();   // s_tok/s_w ready; previous users of s_x/s_act finished

  // Gather x rows -> bf16 LDS: float4 loads, packed 4xbf16 (b64) stores.
  for (int i = tid; i < TILE_M * Hdim / 4; i += 128) {
    const int m = (i * 4) >> 8;
    const int h = (i * 4) & 255;
    const float4 v = *(const float4*)(x + (size_t)s_tok[m] * Hdim + h);
    v4bf o;
    o[0] = (__bf16)v.x; o[1] = (__bf16)v.y;
    o[2] = (__bf16)v.z; o[3] = (__bf16)v.w;
    *(v4bf*)(s_x + m * LDX + h) = o;
  }
  __syncthreads();

  // ---- GEMM1: [32 x 256] @ [256 x 1536] split into g / u, fused SwiGLU ----
  for (int nt = wave; nt < INTER_ / 16; nt += 4) {
    v8f accg0 = {}, accu0 = {};
    v8f accg1 = {}, accu1 = {};
    for (int kk = 0; kk < Hdim; kk += 32) {
      const v16bf a0 = frag_A(s_x + kk, LDX, lane);
      const v16bf a1 = frag_A(s_x + 16 * LDX + kk, LDX, lane);
      const v16bf bg = load_B_global(Wgu_e + (size_t)kk * (2 * INTER_) + nt * 16,
                                     2 * INTER_, lane);
      accg0 = wmma_bf16(a0, bg, accg0);
      accg1 = wmma_bf16(a1, bg, accg1);
      const v16bf bu = load_B_global(Wgu_e + (size_t)kk * (2 * INTER_) + INTER_ + nt * 16,
                                     2 * INTER_, lane);
      accu0 = wmma_bf16(a0, bu, accu0);
      accu1 = wmma_bf16(a1, bu, accu1);
      if (kk + 32 < Hdim)
        __builtin_prefetch(Wgu_e + (size_t)(kk + 32) * (2 * INTER_) + nt * 16, 0, 1);
    }
    const int n = lane & 15, hi = lane >> 4;
#pragma unroll
    for (int r = 0; r < 8; ++r) {
      const int m = r + hi * 8;
      const float g0 = accg0[r], u0 = accu0[r];
      s_act[m * LDA + nt * 16 + n] =
          (__bf16)((g0 / (1.0f + __expf(-g0))) * u0);
      const float g1 = accg1[r], u1 = accu1[r];
      s_act[(m + 16) * LDA + nt * 16 + n] =
          (__bf16)((g1 / (1.0f + __expf(-g1))) * u1);
    }
  }
  __syncthreads();

  // ---- GEMM2: [32 x 768] @ [768 x 256], scale by gate weight, combine ----
  for (int nt = wave; nt < Hdim / 16; nt += 4) {
    v8f acc0 = {}, acc1 = {};
    for (int kk = 0; kk < INTER_; kk += 32) {
      const v16bf a0 = frag_A(s_act + kk, LDA, lane);
      const v16bf a1 = frag_A(s_act + 16 * LDA + kk, LDA, lane);
      const v16bf b  = load_B_global(Wd_e + (size_t)kk * Hdim + nt * 16, Hdim, lane);
      acc0 = wmma_bf16(a0, b, acc0);
      acc1 = wmma_bf16(a1, b, acc1);
      if (kk + 32 < INTER_)
        __builtin_prefetch(Wd_e + (size_t)(kk + 32) * Hdim + nt * 16, 0, 1);
    }
    const int n = lane & 15, hi = lane >> 4;
#pragma unroll
    for (int r = 0; r < 8; ++r) {
      const int m = r + hi * 8;
      const float v0 = acc0[r] * s_w[m];
      const float v1 = acc1[r] * s_w[m + 16];
      float* dst0 = &out[(size_t)s_tok[m] * Hdim + nt * 16 + n];
      float* dst1 = &out[(size_t)s_tok[m + 16] * Hdim + nt * 16 + n];
      if (use_atomic) { atomicAdd(dst0, v0); atomicAdd(dst1, v1); }
      else            { *dst0 += v0;         *dst1 += v1;         }
    }
  }
}

// -------------------- routing: softmax, device-limited top-k, lists ---------
__global__ __launch_bounds__(256)
void moe_route_kernel(const float* __restrict__ x, const float* __restrict__ Wr,
                      int* __restrict__ ecount, float* __restrict__ psum,
                      int* __restrict__ etok, float* __restrict__ ewt) {
  const int t = blockIdx.x;      // token
  const int tid = threadIdx.x;   // 256 threads
  __shared__ float s_xr[Hdim];
  __shared__ float s_l[NEXP];    // logits, later reused as masked scores
  __shared__ float s_s[NEXP];    // softmax scores
  __shared__ float s_red[2];

  s_xr[tid] = x[(size_t)t * Hdim + tid];
  __syncthreads();

  if (tid < NEXP) {
    float acc = 0.0f;
    for (int h = 0; h < Hdim; ++h) acc += s_xr[h] * Wr[h * NEXP + tid];
    s_l[tid] = acc;
  }
  __syncthreads();

  if (tid == 0) {
    float mx = -1e30f;
    for (int e = 0; e < NEXP; ++e) mx = fmaxf(mx, s_l[e]);
    float sm = 0.0f;
    for (int e = 0; e < NEXP; ++e) sm += __expf(s_l[e] - mx);
    s_red[0] = mx; s_red[1] = sm;
  }
  __syncthreads();

  if (tid < NEXP) {
    const float sc = __expf(s_l[tid] - s_red[0]) / s_red[1];
    s_s[tid] = sc;
    atomicAdd(&psum[tid], sc);     // sum over tokens -> P_i * T
  }
  __syncthreads();

  if (tid == 0) {
    // per-device score mass, top-3 devices
    float dsum[NDEV];
    bool  sel[NDEV];
    for (int d = 0; d < NDEV; ++d) {
      float a = 0.0f;
      for (int j = 0; j < EPD_; ++j) a += s_s[d * EPD_ + j];
      dsum[d] = a; sel[d] = false;
    }
    for (int j = 0; j < MAXDEV_; ++j) {
      int best = 0; float bv = -1e30f;
      for (int d = 0; d < NDEV; ++d)
        if (!sel[d] && dsum[d] > bv) { bv = dsum[d]; best = d; }
      sel[best] = true;
    }
    // masked top-6 over scores
    for (int e = 0; e < NEXP; ++e) s_l[e] = sel[e / EPD_] ? s_s[e] : -1e30f;
    int   eidx[TOPK];
    float ew[TOPK];
    for (int k = 0; k < TOPK; ++k) {
      int be = 0; float bv = -1e30f;
      for (int e = 0; e < NEXP; ++e)
        if (s_l[e] > bv) { bv = s_l[e]; be = e; }
      eidx[k] = be; ew[k] = bv; s_l[be] = -1e30f;
    }
    const float mx = ew[0];
    float sm = 0.0f;
    for (int k = 0; k < TOPK; ++k) { ew[k] = __expf(ew[k] - mx); sm += ew[k]; }
    for (int k = 0; k < TOPK; ++k) {
      const int e = eidx[k];
      const int pos = atomicAdd(&ecount[e], 1);
      etok[e * TTOT + pos] = t;
      ewt[e * TTOT + pos]  = ew[k] / sm;
    }
  }
}

// -------------------- aux balance losses ------------------------------------
__global__ void moe_aux_kernel(const int* __restrict__ ecount,
                               const float* __restrict__ psum,
                               float* __restrict__ outsc) {
  if (threadIdx.x != 0 || blockIdx.x != 0) return;
  float eb = 0.0f, db = 0.0f, cb = 0.0f;
  for (int d = 0; d < NDEV; ++d) {
    float df = 0.0f, dP = 0.0f, dc = 0.0f;
    for (int j = 0; j < EPD_; ++j) {
      const int e = d * EPD_ + j;
      const float fi = (float)ecount[e] / ((float)TTOT * TOPK + 1e-10f);
      const float Pi = psum[e] / (float)TTOT;
      eb += fi * Pi;
      df += fi; dP += Pi; dc += (float)ecount[e];
    }
    df /= (float)EPD_;
    db += df * dP;
    cb += (dc / ((float)TTOT * MAXDEV_ + 1e-10f)) * dP;
  }
  eb = fminf(eb * 0.003f, 10.0f);
  db = fminf(db * 0.05f, 10.0f);
  cb = fminf(cb * 0.02f, 10.0f);
  outsc[0] = eb; outsc[1] = db; outsc[2] = cb; outsc[3] = eb + db + cb;
}

// -------------------- routed experts: grouped GEMM over token lists ---------
__global__ __launch_bounds__(128)
void moe_routed_kernel(const float* __restrict__ x, const float* __restrict__ Wgu,
                       const float* __restrict__ Wd, const int* __restrict__ ecount,
                       const int* __restrict__ etok, const float* __restrict__ ewt,
                       float* __restrict__ out) {
  __shared__ __align__(16) __bf16 s_x[TILE_M * LDX];
  __shared__ __align__(16) __bf16 s_act[TILE_M * LDA];
  __shared__ int   s_tok[TILE_M];
  __shared__ float s_w[TILE_M];

  const int e = blockIdx.x;
  const int tile = blockIdx.y;
  const int ct = ecount[e];
  if (tile * TILE_M >= ct) return;

  const int tid = threadIdx.x;
  if (tid < TILE_M) {
    const int i = tile * TILE_M + tid;
    if (i < ct) { s_tok[tid] = etok[e * TTOT + i]; s_w[tid] = ewt[e * TTOT + i]; }
    else        { s_tok[tid] = 0;                  s_w[tid] = 0.0f; }
  }
  expert_tile(x,
              Wgu + (size_t)e * Hdim * (2 * INTER_),
              Wd  + (size_t)e * INTER_ * Hdim,
              out, s_tok, s_w, s_x, s_act, tid, /*atomic=*/true);
}

// -------------------- shared experts (dense, weight 1) ----------------------
__global__ __launch_bounds__(128)
void moe_shared_kernel(const float* __restrict__ x, const float* __restrict__ Wgu_sh,
                       const float* __restrict__ Wd_sh, float* __restrict__ out) {
  __shared__ __align__(16) __bf16 s_x[TILE_M * LDX];
  __shared__ __align__(16) __bf16 s_act[TILE_M * LDA];
  __shared__ int   s_tok[TILE_M];
  __shared__ float s_w[TILE_M];

  const int tid = threadIdx.x;
  if (tid < TILE_M) { s_tok[tid] = blockIdx.x * TILE_M + tid; s_w[tid] = 1.0f; }

  for (int se = 0; se < 2; ++se) {
    expert_tile(x,
                Wgu_sh + (size_t)se * Hdim * (2 * INTER_),
                Wd_sh  + (size_t)se * INTER_ * Hdim,
                out, s_tok, s_w, s_x, s_act, tid, /*atomic=*/false);
  }
}

extern "C" void kernel_launch(void* const* d_in, const int* in_sizes, int n_in,
                              void* d_out, int out_size, void* d_ws, size_t ws_size,
                              hipStream_t stream) {
  const float* x      = (const float*)d_in[0];   // [2,256,256]
  const float* Wr     = (const float*)d_in[1];   // [256,160]
  const float* Wgu_sh = (const float*)d_in[2];   // [2,256,1536]
  const float* Wd_sh  = (const float*)d_in[3];   // [2,768,256]
  const float* Wgu    = (const float*)d_in[4];   // [160,256,1536]
  const float* Wd     = (const float*)d_in[5];   // [160,768,256]
  float* out = (float*)d_out;                    // 131072 output + 4 scalars

  char* ws = (char*)d_ws;
  int*   ecount = (int*)ws;                        // 160 * 4  = 640 B
  float* psum   = (float*)(ws + 640);              // 160 * 4  = 640 B
  int*   etok   = (int*)(ws + 1280);               // 160*512*4
  float* ewt    = (float*)(ws + 1280 + (size_t)NEXP * TTOT * 4);

  hipMemsetAsync(d_ws, 0, 1280, stream);                            // counters
  hipMemsetAsync(d_out, 0, (size_t)TTOT * Hdim * sizeof(float), stream);

  moe_route_kernel<<<TTOT, 256, 0, stream>>>(x, Wr, ecount, psum, etok, ewt);
  moe_aux_kernel<<<1, 32, 0, stream>>>(ecount, psum, out + (size_t)TTOT * Hdim);
  moe_routed_kernel<<<dim3(NEXP, TTOT / TILE_M), 128, 0, stream>>>(x, Wgu, Wd, ecount,
                                                                   etok, ewt, out);
  moe_shared_kernel<<<TTOT / TILE_M, 128, 0, stream>>>(x, Wgu_sh, Wd_sh, out);
}
